// ReformerEncoder_17849884082541
// MI455X (gfx1250) — matrix operand order repaired and checked by reading, not compiled, not run on hardware
//
#include <hip/hip_runtime.h>

// ---------------------------------------------------------------------------
// Types / helpers for CDNA5 (gfx1250) wave32 WMMA + TDM
// ---------------------------------------------------------------------------
typedef __attribute__((ext_vector_type(16))) __bf16 v16bf;
typedef __attribute__((ext_vector_type(8)))  float  v8f;
typedef __attribute__((ext_vector_type(4)))  unsigned int v4u;
typedef __attribute__((ext_vector_type(8)))  int v8i_t;
typedef __attribute__((ext_vector_type(4)))  int v4i_t;

__device__ __forceinline__ unsigned short f2bf(float f) {
  unsigned int u = __float_as_uint(f);
  u += 0x7FFFu + ((u >> 16) & 1u);          // round-to-nearest-even
  return (unsigned short)(u >> 16);
}
__device__ __forceinline__ float bf2f(unsigned short h) {
  return __uint_as_float(((unsigned int)h) << 16);
}

__device__ __forceinline__ v8f wmma_bf16(v16bf a, v16bf b, v8f c) {
  // D = A(16x32) * B(32x16) + C(16x16), f32 accumulate
  return __builtin_amdgcn_wmma_f32_16x16x32_bf16(
      /*neg_a=*/false, a, /*neg_b=*/false, b,
      /*c_mod=*/(short)0, c, /*reuse_a=*/false, /*reuse_b=*/false);
}

// Build a v16bf fragment from two 16-byte aligned 8-ushort chunks.
__device__ __forceinline__ v16bf frag_from2(const unsigned short* p0,
                                            const unsigned short* p1) {
  union { uint4 q[2]; v16bf v; } t;
  t.q[0] = *(const uint4*)p0;
  t.q[1] = *(const uint4*)p1;
  return t.v;
}

// A fragment (16x32 bf16) from a row-major tile (ld in ushorts, multiple of 8).
// Lane layout (ISA 7.12.2): row = lane&15; lanes 0-15 hold K 0..7 & 16..23,
// lanes 16-31 hold K 8..15 & 24..31  ->  two contiguous 8-element runs.
__device__ __forceinline__ v16bf load_a_frag(const unsigned short* base, int ld,
                                             int row0, int k0, int lane) {
  const unsigned short* p =
      base + (row0 + (lane & 15)) * ld + k0 + ((lane >> 4) << 3);
  return frag_from2(p, p + 16);
}

// B fragment (32x16) from an N-major buffer: element (k,n) at baseT[n*ldk + k].
// Lane layout: n = lane&15 fixed; lanes 0-15 hold K 0..7 & 16..23, lanes 16-31 hold K 8..15 & 24..31.
// (Also serves the B = Q^T case with baseT = row-major Q, ldk = row stride.)
__device__ __forceinline__ v16bf load_b_contig(const unsigned short* baseT, int ldk,
                                               int k0, int n0, int lane) {
  const unsigned short* p =
      baseT + (n0 + (lane & 15)) * ldk + k0 + ((lane >> 4) << 3);
  return frag_from2(p, p + 16);
}

// LDS byte address of a shared-memory pointer (generic -> AS(3) cast).
__device__ __forceinline__ unsigned lds_byte_addr(unsigned short* p) {
  return (unsigned)(unsigned long long)(__attribute__((address_space(3))) unsigned short*)p;
}

// ---------------------------------------------------------------------------
// Elementwise conversion kernels
// ---------------------------------------------------------------------------
__global__ void f2bf_kernel(const float* __restrict__ src,
                            unsigned short* __restrict__ dst, size_t n) {
  size_t i = (size_t)blockIdx.x * 256 + threadIdx.x;
  if (i < n) dst[i] = f2bf(src[i]);
}

// x is (16384, 1024); take channels [512,1024) -> bf16 (16384, 512)
__global__ void extract_x2_kernel(const float* __restrict__ x,
                                  unsigned short* __restrict__ x2bf) {
  size_t i = (size_t)blockIdx.x * 256 + threadIdx.x;   // over 16384*512
  size_t tok = i >> 9;
  int    c   = (int)(i & 511);
  x2bf[i] = f2bf(x[tok * 1024 + 512 + c]);
}

// ---------------------------------------------------------------------------
// Generic bf16 GEMM:  C(MxN) = act(A(MxK) @ W(KxN) + bias)
// Block: 256 threads = 8 waves; tile 128x128, K-step 32.
// A tile (128x32 row-major) is DMA'd into LDS by the Tensor Data Mover
// (issued by wave 0, overlapped with the cooperative B-tile transpose).
// B tile is staged N-major so every fragment load is 2x ds_load_b128.
// ---------------------------------------------------------------------------
__global__ void gemm_bf16(const unsigned short* __restrict__ A,
                          const unsigned short* __restrict__ W,
                          const float* __restrict__ bias,
                          unsigned short* __restrict__ Cbf,
                          float* __restrict__ Cf32,
                          int M, int N, int K, int relu) {
  __shared__ unsigned short At[128 * 32];  // row-major (m, k)
  __shared__ unsigned short Bt[128 * 32];  // N-major   (n, k)

  const int tid  = threadIdx.x;
  const int lane = tid & 31;
  const int wave = tid >> 5;
  const int bm = blockIdx.y * 128;
  const int bn = blockIdx.x * 128;

  const unsigned at_lds = lds_byte_addr(At);

  v8f acc[8];
#pragma unroll
  for (int j = 0; j < 8; ++j) acc[j] = (v8f){0.f, 0.f, 0.f, 0.f, 0.f, 0.f, 0.f, 0.f};

  for (int k0 = 0; k0 < K; k0 += 32) {
    if (wave == 0) {
      // Tensor DMA: 2D tile, 128 rows x 32 cols of 2-byte elements,
      // row stride = K elements (ISA 8.3/8.4 D# layout).
      unsigned long long ga =
          (unsigned long long)(const char*)(A + (size_t)bm * K + k0);
      v4u g0;
      g0.x = 1u;                                    // count=1 (valid), user mode
      g0.y = at_lds;                                // lds_addr (bytes)
      g0.z = (unsigned)(ga & 0xFFFFFFFFu);          // global_addr[31:0]
      g0.w = (unsigned)((ga >> 32) & 0x01FFFFFFu)   // global_addr[56:32]
             | 0x80000000u;                         // type = 2 ("image")
      v8i_t g1;
      g1[0] = (1 << 16);        // workgroup_mask=0, data_size=1 (2 bytes)
      g1[1] = (32 << 16);       // tensor_dim0[15:0] = 32
      g1[2] = (128 << 16);      // tensor_dim0 hi=0 | tensor_dim1 lo = 128
      g1[3] = (32 << 16);       // tensor_dim1 hi=0 | tile_dim0 = 32
      g1[4] = 128;              // tile_dim1 = 128, tile_dim2 = 0
      g1[5] = K;                // tensor_dim0_stride[31:0] = K elements
      g1[6] = 0;                // stride hi / tensor_dim1_stride lo
      g1[7] = 0;
      v4i_t gz4 = (v4i_t){0, 0, 0, 0};
      v8i_t gz8 = (v8i_t){0, 0, 0, 0, 0, 0, 0, 0};
      __builtin_amdgcn_tensor_load_to_lds(g0, g1, gz4, gz4, gz8, 0);
    }

    // All waves (incl. wave0) stage B: read W row-major (coalesced along n),
    // store N-major transposed.  Overlaps with the in-flight TDM transfer.
    for (int i = tid; i < 32 * 128; i += 256) {
      int r = i >> 7, c = i & 127;     // r = k within tile, c = n within tile
      Bt[c * 32 + r] = W[(size_t)(k0 + r) * N + bn + c];
    }
    if (k0 + 32 < K) {  // hint L2 for the next K tile -> global_prefetch_b8
      __builtin_prefetch(W + (size_t)(k0 + 32 + (tid >> 3)) * N + bn, 0, 1);
    }
    if (wave == 0) __builtin_amdgcn_s_wait_tensorcnt(0);
    __syncthreads();

    // Pre-load A fragment + all 8 B fragments, then issue 8 WMMAs back-to-back.
    v16bf a = load_a_frag(At, 32, wave * 16, 0, lane);
    v16bf bfrag[8];
#pragma unroll
    for (int j = 0; j < 8; ++j) bfrag[j] = load_b_contig(Bt, 32, 0, j * 16, lane);
#pragma unroll
    for (int j = 0; j < 8; ++j) acc[j] = wmma_bf16(a, bfrag[j], acc[j]);
    __syncthreads();
  }

  // Epilogue: C lane layout: n = lane&15, m = vgpr + 8*(lane>>4)
  const int mh = lane >> 4;
  const int nl = lane & 15;
#pragma unroll
  for (int j = 0; j < 8; ++j) {
    int gn = bn + j * 16 + nl;
    float bv = bias ? bias[gn] : 0.f;
#pragma unroll
    for (int r = 0; r < 8; ++r) {
      int gm = bm + wave * 16 + r + 8 * mh;
      float v = acc[j][r] + bv;
      if (relu) v = v > 0.f ? v : 0.f;
      size_t o = (size_t)gm * N + gn;
      if (Cbf)  Cbf[o]  = f2bf(v);
      if (Cf32) Cf32[o] = v;
    }
  }
}

// ---------------------------------------------------------------------------
// LSH hashing: proj = qk @ H (512x32); bucket = argmax([proj, -proj])
// One thread per token; H staged in 64KB LDS; qk rows read as uint4.
// ---------------------------------------------------------------------------
__global__ void hash_kernel(const unsigned short* __restrict__ QKbf,
                            const float* __restrict__ H,
                            int* __restrict__ hashes) {
  extern __shared__ char smraw[];
  float* Hs = (float*)smraw;  // 512*32 floats
  for (int i = threadIdx.x; i < 512 * 32; i += 256) Hs[i] = H[i];
  __syncthreads();

  int tok = blockIdx.x * 256 + threadIdx.x;  // 16384 tokens
  const unsigned short* q = QKbf + (size_t)tok * 512;

  float proj[32];
#pragma unroll
  for (int c = 0; c < 32; ++c) proj[c] = 0.f;

  for (int k8 = 0; k8 < 512; k8 += 8) {
    uint4 qq = *(const uint4*)(q + k8);
    unsigned int w[4] = {qq.x, qq.y, qq.z, qq.w};
#pragma unroll
    for (int t = 0; t < 4; ++t) {
      float q0 = bf2f((unsigned short)(w[t] & 0xFFFFu));
      float q1 = bf2f((unsigned short)(w[t] >> 16));
      const float* hr = Hs + (k8 + 2 * t) * 32;
#pragma unroll
      for (int c = 0; c < 32; ++c) proj[c] += q0 * hr[c] + q1 * hr[32 + c];
    }
  }
  float best = proj[0];
  int   bi   = 0;
  for (int c = 1; c < 32; ++c)
    if (proj[c] > best) { best = proj[c]; bi = c; }
  for (int c = 0; c < 32; ++c)
    if (-proj[c] > best) { best = -proj[c]; bi = 32 + c; }
  hashes[tok] = bi;
}

// ---------------------------------------------------------------------------
// Stable argsort of hashes per batch (64 buckets, 4096 tokens): O(n^2) rank.
// idx[d*4096 + pos] = original token index (matches jnp.argsort stable order).
// ---------------------------------------------------------------------------
__global__ void rank_sort_kernel(const int* __restrict__ hashes,
                                 int* __restrict__ idx) {
  int d = blockIdx.y;
  int i = blockIdx.x * 256 + threadIdx.x;  // 0..4095
  const int* h = hashes + d * 4096;
  int hi  = h[i];
  int pos = 0;
  for (int j = 0; j < 4096; ++j) {
    int hj = h[j];
    pos += (hj < hi) | ((hj == hi) & (j < i));
  }
  idx[d * 4096 + pos] = i;
}

// ---------------------------------------------------------------------------
// Fused per-chunk LSH attention. One workgroup (8 waves) per chunk of 128
// sorted tokens.  LDS: Q row-major (128x512), V channel-major (512x128),
// per-wave S strips -> 288KB of the 320KB WGP LDS.
//   S = (Q Q^T)/8, diagonal / PENALTY;  out = S @ V, scattered back via idx.
// ---------------------------------------------------------------------------
__global__ void lsh_attention_kernel(const unsigned short* __restrict__ QKbf,
                                     const unsigned short* __restrict__ Vbf,
                                     const int* __restrict__ idx,
                                     unsigned short* __restrict__ ATTNbf) {
  extern __shared__ unsigned short sm[];
  unsigned short* Qs = sm;                   // [128][512]  (key row, k)
  unsigned short* Vt = sm + 128 * 512;       // [512][128]  (channel, key)
  unsigned short* Ss = sm + 2 * 128 * 512;   // 8 waves * [16][128]

  const int b    = blockIdx.x;     // 0..127 chunks
  const int d    = b >> 5;         // batch
  const int cpos = b & 31;         // chunk within batch
  const int* idxd = idx + d * 4096 + cpos * 128;

  const int tid  = threadIdx.x;
  const int lane = tid & 31;
  const int wave = tid >> 5;

  // Gather 128 rows; Q stays row-major, V is transposed to channel-major.
  for (int u = tid; u < 128 * 256; u += 256) {
    int r  = u >> 8;       // key row 0..127
    int c2 = u & 255;      // uint index within row (256 uints = 512 bf16)
    int tok = idxd[r];
    size_t src = ((size_t)(d * 4096 + tok) * 512 + (size_t)c2 * 2);
    ((unsigned int*)Qs)[r * 256 + c2] = *(const unsigned int*)(QKbf + src);
    unsigned int vv = *(const unsigned int*)(Vbf + src);
    Vt[(2 * c2)     * 128 + r] = (unsigned short)(vv & 0xFFFFu);
    Vt[(2 * c2 + 1) * 128 + r] = (unsigned short)(vv >> 16);
  }
  __syncthreads();

  unsigned short* Sw   = Ss + wave * 16 * 128;
  const int       row0 = wave * 16;
  const int mh = lane >> 4, nl = lane & 15;

  // Phase 1: S strip (16 x 128) = Q_strip @ Q^T, scale + diagonal penalty.
#pragma unroll 1
  for (int j = 0; j < 8; ++j) {
    v8f acc = (v8f){0.f, 0.f, 0.f, 0.f, 0.f, 0.f, 0.f, 0.f};
#pragma unroll
    for (int k0 = 0; k0 < 512; k0 += 32) {
      v16bf a  = load_a_frag(Qs, 512, row0, k0, lane);
      v16bf bq = load_b_contig(Qs, 512, k0, j * 16, lane);  // B = Q^T
      acc = wmma_bf16(a, bq, acc);
    }
#pragma unroll
    for (int r = 0; r < 8; ++r) {
      int m = r + 8 * mh;                    // query row within strip
      int n = j * 16 + nl;                   // key index within chunk
      float v = acc[r] * 0.125f;             // / sqrt(DQK)
      if (row0 + m == n) v *= 1.0f / 100000.0f;  // / PENALTY on diagonal
      Sw[m * 128 + n] = f2bf(v);
    }
  }

  // Phase 2: out strip (16 x 512) = S_strip @ V, scatter rows via idx.
#pragma unroll 1
  for (int j = 0; j < 32; ++j) {
    v8f acc = (v8f){0.f, 0.f, 0.f, 0.f, 0.f, 0.f, 0.f, 0.f};
#pragma unroll
    for (int k0 = 0; k0 < 128; k0 += 32) {
      v16bf a  = load_a_frag(Sw, 128, 0, k0, lane);
      v16bf bv = load_b_contig(Vt, 128, k0, j * 16, lane);
      acc = wmma_bf16(a, bv, acc);
    }
#pragma unroll
    for (int r = 0; r < 8; ++r) {
      int m   = r + 8 * mh;
      int tok = idxd[row0 + m];
      ATTNbf[((size_t)(d * 4096 + tok)) * 512 + j * 16 + nl] = f2bf(acc[r]);
    }
  }
}

// ---------------------------------------------------------------------------
// Residual + LayerNorm over 512 channels; writes into one half of the
// (.., 1024) output row, optionally also a bf16 copy for the next GEMM.
// One block (256 threads) per token.
// ---------------------------------------------------------------------------
__global__ void residual_ln_kernel(const float* __restrict__ gin,   // (tok,512)
                                   const float* __restrict__ xfull, // (tok,1024)
                                   int xoff,
                                   const float* __restrict__ gamma,
                                   const float* __restrict__ beta,
                                   float* __restrict__ outfull, int outoff,
                                   unsigned short* __restrict__ outbf) {
  __shared__ float buf[256];
  const int tok = blockIdx.x;
  const int tid = threadIdx.x;
  const float* xr = xfull + (size_t)tok * 1024 + xoff;
  const float* gr = gin + (size_t)tok * 512;

  float v0 = gr[tid]       + xr[tid];
  float v1 = gr[tid + 256] + xr[tid + 256];

  buf[tid] = v0 + v1;
  __syncthreads();
  for (int s = 128; s > 0; s >>= 1) {
    if (tid < s) buf[tid] += buf[tid + s];
    __syncthreads();
  }
  float mu = buf[0] * (1.0f / 512.0f);
  __syncthreads();

  float d0 = v0 - mu, d1 = v1 - mu;
  buf[tid] = d0 * d0 + d1 * d1;
  __syncthreads();
  for (int s = 128; s > 0; s >>= 1) {
    if (tid < s) buf[tid] += buf[tid + s];
    __syncthreads();
  }
  float inv = rsqrtf(buf[0] * (1.0f / 512.0f) + 1e-5f);

  float y0 = d0 * inv * gamma[tid]       + beta[tid];
  float y1 = d1 * inv * gamma[tid + 256] + beta[tid + 256];

  float* orow = outfull + (size_t)tok * 1024 + outoff;
  orow[tid]       = y0;
  orow[tid + 256] = y1;
  if (outbf) {
    unsigned short* ob = outbf + (size_t)tok * 512;
    ob[tid]       = f2bf(y0);
    ob[tid + 256] = f2bf(y1);
  }
}

// ---------------------------------------------------------------------------
// Host-side orchestration
// ---------------------------------------------------------------------------
extern "C" void kernel_launch(void* const* d_in, const int* in_sizes, int n_in,
                              void* d_out, int out_size, void* d_ws, size_t ws_size,
                              hipStream_t stream) {
  (void)in_sizes; (void)n_in; (void)out_size; (void)ws_size;

  const float* x       = (const float*)d_in[0];   // (4,4096,1024)
  const float* Wqk_w   = (const float*)d_in[1];   // (512,512)
  const float* Wqk_b   = (const float*)d_in[2];
  const float* Wv_w    = (const float*)d_in[3];
  const float* Wv_b    = (const float*)d_in[4];
  const float* unify_w = (const float*)d_in[5];
  const float* unify_b = (const float*)d_in[6];
  const float* hashM   = (const float*)d_in[7];   // (512,32)
  const float* ln1_g   = (const float*)d_in[8];
  const float* ln1_b   = (const float*)d_in[9];
  const float* ff1_w   = (const float*)d_in[10];  // (512,2048)
  const float* ff1_b   = (const float*)d_in[11];
  const float* ff2_w   = (const float*)d_in[12];  // (2048,512)
  const float* ff2_b   = (const float*)d_in[13];
  const float* ln2_g   = (const float*)d_in[14];
  const float* ln2_b   = (const float*)d_in[15];
  float* out = (float*)d_out;

  const size_t MTOT = (size_t)4 * 4096;           // 16384 tokens

  // Workspace carve-up (ushort units)
  unsigned short* wsu = (unsigned short*)d_ws;
  size_t off = 0;
  unsigned short* Wqkb = wsu + off; off += (size_t)512 * 512;
  unsigned short* Wvb  = wsu + off; off += (size_t)512 * 512;
  unsigned short* Wub  = wsu + off; off += (size_t)512 * 512;
  unsigned short* Wf1b = wsu + off; off += (size_t)512 * 2048;
  unsigned short* Wf2b = wsu + off; off += (size_t)2048 * 512;
  unsigned short* X2b  = wsu + off; off += MTOT * 512;
  unsigned short* QKb  = wsu + off; off += MTOT * 512;
  unsigned short* Vb   = wsu + off; off += MTOT * 512;
  unsigned short* ATb  = wsu + off; off += MTOT * 512;
  unsigned short* Y1b  = wsu + off; off += MTOT * 512;
  unsigned short* FFb  = wsu + off; off += (size_t)4096 * 2048;   // one M-chunk
  float* F32t   = (float*)(wsu + off); off += 2 * MTOT * 512;     // f32 scratch
  int*   hashes = (int*)(wsu + off);   off += 2 * 16384;
  int*   idxbuf = (int*)(wsu + off);   off += 2 * 16384;

  const size_t ATTN_LDS = (size_t)(2 * 128 * 512 + 8 * 16 * 128) * sizeof(unsigned short);
  const size_t HASH_LDS = (size_t)512 * 32 * sizeof(float);

  auto cvt = [&](const float* s, unsigned short* dst, size_t n) {
    f2bf_kernel<<<dim3((unsigned)((n + 255) / 256)), dim3(256), 0, stream>>>(s, dst, n);
  };

  // 1) weights -> bf16, x2 slice -> bf16
  cvt(Wqk_w, Wqkb, (size_t)512 * 512);
  cvt(Wv_w,  Wvb,  (size_t)512 * 512);
  cvt(unify_w, Wub, (size_t)512 * 512);
  cvt(ff1_w, Wf1b, (size_t)512 * 2048);
  cvt(ff2_w, Wf2b, (size_t)2048 * 512);
  extract_x2_kernel<<<dim3((unsigned)(MTOT * 512 / 256)), dim3(256), 0, stream>>>(x, X2b);

  // 2) qk = x2 @ Wqk + b   (bf16 out)
  gemm_bf16<<<dim3(512 / 128, (unsigned)(MTOT / 128)), dim3(256), 0, stream>>>(
      X2b, Wqkb, Wqk_b, QKb, nullptr, (int)MTOT, 512, 512, 0);

  // 3) LSH hashes + stable sort
  hash_kernel<<<dim3(64), dim3(256), HASH_LDS, stream>>>(QKb, hashM, hashes);
  rank_sort_kernel<<<dim3(16, 4), dim3(256), 0, stream>>>(hashes, idxbuf);

  // 4) v = x2 @ Wv + b
  gemm_bf16<<<dim3(512 / 128, (unsigned)(MTOT / 128)), dim3(256), 0, stream>>>(
      X2b, Wvb, Wv_b, Vb, nullptr, (int)MTOT, 512, 512, 0);

  // 5) chunked attention (128 chunks)
  lsh_attention_kernel<<<dim3(128), dim3(256), ATTN_LDS, stream>>>(QKb, Vb, idxbuf, ATb);

  // 6) unify GEMM (f32 out), then y1 = LN(x1 + attn) -> out[:, :512] and bf16
  gemm_bf16<<<dim3(512 / 128, (unsigned)(MTOT / 128)), dim3(256), 0, stream>>>(
      ATb, Wub, unify_b, nullptr, F32t, (int)MTOT, 512, 512, 0);
  residual_ln_kernel<<<dim3((unsigned)MTOT), dim3(256), 0, stream>>>(
      F32t, x, 0, ln1_g, ln1_b, out, 0, Y1b);

  // 7) FFN in 4 M-chunks of 4096 rows: relu(y1@ff1+b) @ ff2 + b -> F32t
  for (int mc = 0; mc < 4; ++mc) {
    const size_t ro = (size_t)mc * 4096;
    gemm_bf16<<<dim3(2048 / 128, 4096 / 128), dim3(256), 0, stream>>>(
        Y1b + ro * 512, Wf1b, ff1_b, FFb, nullptr, 4096, 2048, 512, 1);
    gemm_bf16<<<dim3(512 / 128, 4096 / 128), dim3(256), 0, stream>>>(
        FFb, Wf2b, ff2_b, nullptr, F32t + ro * 512, 4096, 512, 2048, 0);
  }

  // 8) y2 = LN(x2 + ff) -> out[:, 512:]
  residual_ln_kernel<<<dim3((unsigned)MTOT), dim3(256), 0, stream>>>(
      F32t, x, 512, ln2_g, ln2_b, out, 512, nullptr);
}
